// MultiHeadAttention_48962627175194
// MI455X (gfx1250) — matrix operand order repaired and checked by reading, not compiled, not run on hardware
//
#include <hip/hip_runtime.h>

typedef __attribute__((ext_vector_type(16))) __bf16 v16bf;
typedef __attribute__((ext_vector_type(8)))  float  v8f;

#define B_  4
#define S_  2048
#define D_  1024
#define H_  16
#define DK_ 64

// ---------- helpers ----------
__device__ __forceinline__ unsigned short f2bf(float f) {
  unsigned int u = __float_as_uint(f);
  u += 0x7fffu + ((u >> 16) & 1u);   // round-to-nearest-even
  return (unsigned short)(u >> 16);
}

// A-fragment (16x32 bf16, M x K): lane m = lane&15,
// element e -> K = (lane>=16 ? 16 : 0) + e (contiguous, matches B-fragment).
// Two 16B-contiguous chunks -> compiler merges into 2x global_load_b128.
__device__ __forceinline__ v16bf load_a_frag(const unsigned short* row_base, int hi8) {
  union { unsigned int u[8]; v16bf v; } r;
  constexpr int kof[8] = {0, 2, 4, 6, 8, 10, 12, 14};
#pragma unroll
  for (int j = 0; j < 8; ++j)
    r.u[j] = *(const unsigned int*)(row_base + kof[j] + 2 * hi8);
  return r.v;
}

// A-fragment from LDS row of 64 bf16 (P staging), kc selects keys 0..31 / 32..63
__device__ __forceinline__ v16bf lds_a_frag(const unsigned short* row, int hi, int kc) {
  union { unsigned int u[8]; v16bf v; } r;
  const unsigned int* prow = (const unsigned int*)row;
  constexpr int kof[8] = {0, 1, 2, 3, 4, 5, 6, 7};
#pragma unroll
  for (int j = 0; j < 8; ++j) r.u[j] = prow[kof[j] + hi * 8 + kc * 16];
  return r.v;
}

// B-fragment (32x16 bf16, K x N): lane n = lane&15, element e -> K = hi*16 + e,
// i.e. 16 contiguous bf16 (32B) per lane when memory is contiguous along K.
__device__ __forceinline__ v16bf load_b_frag(const unsigned short* p) {
  return *(const v16bf*)p;   // 32B-aligned by construction
}

__device__ __forceinline__ v8f wmma_bf16(v16bf a, v16bf b, v8f c) {
  return __builtin_amdgcn_wmma_f32_16x16x32_bf16(false, a, false, b, (short)0, c, false, false);
}

// ---------- conversion kernels ----------
__global__ void cvt_bf16(const float* __restrict__ src, unsigned short* __restrict__ dst, int n) {
  int i = blockIdx.x * blockDim.x + threadIdx.x;
  if (i < n) dst[i] = f2bf(src[i]);
}

// Wq/Wk/Wv: src [H=16][D=1024][64]  ->  dst [(h*64+c)][1024]  (output-major, contiguous in d)
__global__ void cvt_wqkv(const float* __restrict__ src, unsigned short* __restrict__ dst) {
  int i = blockIdx.x * blockDim.x + threadIdx.x;   // 0 .. 1048575
  int h = i >> 16, r = i & 65535, d = r >> 6, c = r & 63;
  dst[(size_t)((h << 6) + c) * 1024 + d] = f2bf(src[i]);
}

// Wo: src [D][D] (input-major) -> dst [n][d]
__global__ void cvt_wot(const float* __restrict__ src, unsigned short* __restrict__ dst) {
  int i = blockIdx.x * blockDim.x + threadIdx.x;   // 0 .. 1048575
  int d = i >> 10, nn = i & 1023;
  dst[(size_t)nn * 1024 + d] = f2bf(src[i]);
}

// ---------- QKV projection: [8192x1024] x [1024x3072], 32x64 tile per wave ----------
__global__ void __launch_bounds__(256) qkv_gemm(const unsigned short* __restrict__ X,
                                                const unsigned short* __restrict__ Wt,
                                                unsigned short* __restrict__ Q,
                                                unsigned short* __restrict__ K,
                                                unsigned short* __restrict__ V) {
  const int NTB = 48;                          // 3072/64 column blocks
  int lane = threadIdx.x & 31, wib = threadIdx.x >> 5;
  int tile = blockIdx.x * 8 + wib;
  int ntb = tile % NTB, mtb = tile / NTB;      // mtb: 0..255 (32 rows each)
  int n = lane & 15, hi = lane >> 4, hi8 = hi * 8, m = n;

  const unsigned short* arow0 = X + (size_t)(mtb * 32 + m) * 1024;
  const unsigned short* arow1 = arow0 + (size_t)16 * 1024;
  const unsigned short* brow0 = Wt + (size_t)(ntb * 64 + n) * 1024 + hi * 16;

  v8f acc[2][4];
#pragma unroll
  for (int mi = 0; mi < 2; ++mi)
#pragma unroll
    for (int ni = 0; ni < 4; ++ni) acc[mi][ni] = (v8f){};

#pragma unroll 2
  for (int k0 = 0; k0 < 1024; k0 += 32) {
    v16bf a0 = load_a_frag(arow0 + k0, hi8);
    v16bf a1 = load_a_frag(arow1 + k0, hi8);
#pragma unroll
    for (int ni = 0; ni < 4; ++ni) {
      v16bf b = load_b_frag(brow0 + (size_t)ni * 16 * 1024 + k0);
      acc[0][ni] = wmma_bf16(a0, b, acc[0][ni]);
      acc[1][ni] = wmma_bf16(a1, b, acc[1][ni]);
    }
  }

  // ---- epilogue: wave-uniform destination + affine strides (32 | 2048 => b_ uniform) ----
  int cg = ntb * 64, qkv = cg >> 10, h = (cg & 1023) >> 6;   // 64-col block == one head slice
  int token0 = mtb * 32;
  int b_ = token0 >> 11, s0 = token0 & 2047;
  size_t bh = (size_t)(b_ * H_ + h);
  unsigned short* dst;
  size_t tokStride, colStride;
  if (qkv == 0)      { dst = Q + (bh * S_ + s0) * 64;  tokStride = 64; colStride = 1;  }
  else if (qkv == 1) { dst = K + (bh * S_ + s0) * 64;  tokStride = 64; colStride = 1;  }
  else               { dst = V + bh * 64 * S_ + s0;    tokStride = 1;  colStride = S_; }
#pragma unroll
  for (int mi = 0; mi < 2; ++mi)
#pragma unroll
    for (int ni = 0; ni < 4; ++ni)
#pragma unroll
      for (int v = 0; v < 8; ++v) {
        int tl = mi * 16 + v + hi8;
        dst[(size_t)tl * tokStride + (size_t)(ni * 16 + n) * colStride] = f2bf(acc[mi][ni][v]);
      }
}

// ---------- causal flash attention: one wave per (b,h, 16-query tile), 64-key chunks ----------
__global__ void __launch_bounds__(256) attn(const unsigned short* __restrict__ Q,
                                            const unsigned short* __restrict__ Kb,
                                            const unsigned short* __restrict__ Vt,
                                            unsigned short* __restrict__ O) {
  __shared__ unsigned short Plds[8][16][64];   // per-wave 2KB P staging (D-layout -> A-layout)
  int lane = threadIdx.x & 31, wib = threadIdx.x >> 5;
  int w = blockIdx.x * 8 + wib;
  int bh = w >> 7, qt = w & 127, q0 = qt * 16;
  int n = lane & 15, hi = lane >> 4, hi8 = hi * 8, m = n;
  size_t bhS = (size_t)bh * S_;

  const unsigned short* qrow = Q + (bhS + q0 + m) * 64;
  v16bf qa0 = load_a_frag(qrow, hi8);          // dims 0..31
  v16bf qa1 = load_a_frag(qrow + 32, hi8);     // dims 32..63

  float mi[8], li[8];
  v8f o[4];
#pragma unroll
  for (int t = 0; t < 4; ++t) o[t] = (v8f){};
#pragma unroll
  for (int v = 0; v < 8; ++v) { mi[v] = -3.0e38f; li[v] = 0.f; }

  int nch = (q0 + 16 + 63) >> 6;               // 64-key chunks, causal bound
  for (int c = 0; c < nch; ++c) {
    int k0 = c * 64;
    // ---- S = Q * K^T for four 16-key tiles (K-dim 64 via 2 WMMAs each) ----
    v8f s[4];
#pragma unroll
    for (int t = 0; t < 4; ++t) {
      const unsigned short* kr = Kb + (bhS + k0 + t * 16 + n) * 64 + hi * 16;
      s[t] = (v8f){};
      s[t] = wmma_bf16(qa0, load_b_frag(kr),      s[t]);
      s[t] = wmma_bf16(qa1, load_b_frag(kr + 32), s[t]);
    }

    // ---- scale, causal mask, online softmax (one reduction round per 64 keys) ----
#pragma unroll
    for (int v = 0; v < 8; ++v) {
      int qi = q0 + v + hi8;
      float a0 = (k0 + n      > qi) ? -3.0e38f : s[0][v] * 0.125f;
      float a1 = (k0 + 16 + n > qi) ? -3.0e38f : s[1][v] * 0.125f;
      float a2 = (k0 + 32 + n > qi) ? -3.0e38f : s[2][v] * 0.125f;
      float a3 = (k0 + 48 + n > qi) ? -3.0e38f : s[3][v] * 0.125f;
      float lm = fmaxf(fmaxf(a0, a1), fmaxf(a2, a3));
      lm = fmaxf(lm, __shfl_xor(lm, 1, 32));
      lm = fmaxf(lm, __shfl_xor(lm, 2, 32));
      lm = fmaxf(lm, __shfl_xor(lm, 4, 32));
      lm = fmaxf(lm, __shfl_xor(lm, 8, 32));
      float mnew  = fmaxf(mi[v], lm);
      float alpha = __expf(mi[v] - mnew);
      float p0 = __expf(a0 - mnew), p1 = __expf(a1 - mnew);
      float p2 = __expf(a2 - mnew), p3 = __expf(a3 - mnew);
      float rs = (p0 + p1) + (p2 + p3);
      rs += __shfl_xor(rs, 1, 32);
      rs += __shfl_xor(rs, 2, 32);
      rs += __shfl_xor(rs, 4, 32);
      rs += __shfl_xor(rs, 8, 32);
      li[v] = li[v] * alpha + rs;
      mi[v] = mnew;
      o[0][v] *= alpha; o[1][v] *= alpha; o[2][v] *= alpha; o[3][v] *= alpha;
      int M = v + hi8;
      Plds[wib][M][n]      = f2bf(p0);
      Plds[wib][M][16 + n] = f2bf(p1);
      Plds[wib][M][32 + n] = f2bf(p2);
      Plds[wib][M][48 + n] = f2bf(p3);
    }

    // ---- P A-fragments from LDS (same-wave in-order DS, no barrier) ----
    v16bf pa0 = lds_a_frag(&Plds[wib][m][0], hi, 0);   // keys 0..31
    v16bf pa1 = lds_a_frag(&Plds[wib][m][0], hi, 1);   // keys 32..63

    // ---- O += P * V   (V transposed -> contiguous B-fragments) ----
    const unsigned short* vb = Vt + ((size_t)bh * 64 + n) * S_ + k0 + hi * 16;
#pragma unroll
    for (int nt = 0; nt < 4; ++nt) {
      const unsigned short* vrow = vb + (size_t)nt * 16 * S_;
      o[nt] = wmma_bf16(pa0, load_b_frag(vrow),      o[nt]);
      o[nt] = wmma_bf16(pa1, load_b_frag(vrow + 32), o[nt]);
    }
  }

  // ---- finalize: O /= l, write concat-head bf16 [B,S,1024] ----
  int b_ = bh >> 4, h = bh & 15;
  unsigned short* obase = O + ((size_t)(b_ * S_ + q0)) * 1024 + h * 64 + n;
#pragma unroll
  for (int v = 0; v < 8; ++v) {
    float inv = 1.0f / li[v];
    size_t base = (size_t)(v + hi8) * 1024;
    obase[base]      = f2bf(o[0][v] * inv);
    obase[base + 16] = f2bf(o[1][v] * inv);
    obase[base + 32] = f2bf(o[2][v] * inv);
    obase[base + 48] = f2bf(o[3][v] * inv);
  }
}

// ---------- output projection: [8192x1024] x [1024x1024] + bias, 32x64 tile per wave ----------
__global__ void __launch_bounds__(256) out_proj(const unsigned short* __restrict__ Oin,
                                                const unsigned short* __restrict__ Wot,
                                                const float* __restrict__ bo,
                                                float* __restrict__ out) {
  const int NTB = 16;                          // 1024/64 column blocks
  int lane = threadIdx.x & 31, wib = threadIdx.x >> 5;
  int tile = blockIdx.x * 8 + wib;
  int ntb = tile % NTB, mtb = tile / NTB;
  int n = lane & 15, hi = lane >> 4, hi8 = hi * 8, m = n;

  const unsigned short* arow0 = Oin + (size_t)(mtb * 32 + m) * 1024;
  const unsigned short* arow1 = arow0 + (size_t)16 * 1024;
  const unsigned short* brow0 = Wot + (size_t)(ntb * 64 + n) * 1024 + hi * 16;

  v8f acc[2][4];
#pragma unroll
  for (int mi = 0; mi < 2; ++mi)
#pragma unroll
    for (int ni = 0; ni < 4; ++ni) acc[mi][ni] = (v8f){};

#pragma unroll 2
  for (int k0 = 0; k0 < 1024; k0 += 32) {
    v16bf a0 = load_a_frag(arow0 + k0, hi8);
    v16bf a1 = load_a_frag(arow1 + k0, hi8);
#pragma unroll
    for (int ni = 0; ni < 4; ++ni) {
      v16bf b = load_b_frag(brow0 + (size_t)ni * 16 * 1024 + k0);
      acc[0][ni] = wmma_bf16(a0, b, acc[0][ni]);
      acc[1][ni] = wmma_bf16(a1, b, acc[1][ni]);
    }
  }

  float* obase = out + (size_t)(mtb * 32) * 1024 + ntb * 64 + n;
#pragma unroll
  for (int ni = 0; ni < 4; ++ni) {
    float bias = bo[ntb * 64 + ni * 16 + n];
#pragma unroll
    for (int mi = 0; mi < 2; ++mi)
#pragma unroll
      for (int v = 0; v < 8; ++v)
        obase[(size_t)(mi * 16 + v + hi8) * 1024 + ni * 16] = acc[mi][ni][v] + bias;
  }
}

// ---------- host ----------
extern "C" void kernel_launch(void* const* d_in, const int* in_sizes, int n_in,
                              void* d_out, int out_size, void* d_ws, size_t ws_size,
                              hipStream_t stream) {
  const float* emb = (const float*)d_in[0];
  const float* Wq  = (const float*)d_in[1];
  const float* Wk  = (const float*)d_in[2];
  const float* Wv  = (const float*)d_in[3];
  const float* Wo  = (const float*)d_in[4];
  const float* bo  = (const float*)d_in[5];

  const size_t NX = (size_t)B_ * S_ * D_;      // 8,388,608
  const size_t NW = 1048576;                   // 1024*1024
  char* ws = (char*)d_ws;
  unsigned short* Xbf = (unsigned short*)ws; ws += NX * 2;
  unsigned short* Wt  = (unsigned short*)ws; ws += 3 * NW * 2;
  unsigned short* Wot = (unsigned short*)ws; ws += NW * 2;
  unsigned short* Qbf = (unsigned short*)ws; ws += NX * 2;
  unsigned short* Kbf = (unsigned short*)ws; ws += NX * 2;
  unsigned short* Vtb = (unsigned short*)ws; ws += NX * 2;
  unsigned short* Obf = (unsigned short*)ws; ws += NX * 2;

  hipLaunchKernelGGL(cvt_bf16, dim3((unsigned)(NX / 256)), dim3(256), 0, stream, emb, Xbf, (int)NX);
  hipLaunchKernelGGL(cvt_wqkv, dim3(NW / 256), dim3(256), 0, stream, Wq, Wt);
  hipLaunchKernelGGL(cvt_wqkv, dim3(NW / 256), dim3(256), 0, stream, Wk, Wt + NW);
  hipLaunchKernelGGL(cvt_wqkv, dim3(NW / 256), dim3(256), 0, stream, Wv, Wt + 2 * NW);
  hipLaunchKernelGGL(cvt_wot,  dim3(NW / 256), dim3(256), 0, stream, Wo, Wot);

  hipLaunchKernelGGL(qkv_gemm, dim3(256 * 48 / 8), dim3(256), 0, stream, Xbf, Wt, Qbf, Kbf, Vtb);
  hipLaunchKernelGGL(attn,     dim3(8192 / 8),     dim3(256), 0, stream, Qbf, Kbf, Vtb, Obf);
  hipLaunchKernelGGL(out_proj, dim3(256 * 16 / 8), dim3(256), 0, stream, Obf, Wot, bo, (float*)d_out);
}
